// MultiHeadSlotAttention_42013370090144
// MI455X (gfx1250) — compile-verified, matrix operand order checked by reading, not run on hardware
//
#include <hip/hip_runtime.h>
#include <stdint.h>

// ---------------------------------------------------------------- constants
#define BB   32
#define NN   4096
#define DD   512
#define HH   8
#define DH   64
#define NS   16
#define HID  2048
#define SCALE 0.044194173824159216f   // 512^-0.5

typedef __attribute__((ext_vector_type(16))) __bf16 v16bf;
typedef __attribute__((ext_vector_type(8)))  float  v8f;

union FragAB { v16bf v; uint4 q[2]; unsigned short s[16]; };

__device__ inline unsigned short f32_to_bf16(float f) {
  union { float f; uint32_t u; } c; c.f = f;
  uint32_t u = c.u;
  uint32_t r = u + 0x7fffu + ((u >> 16) & 1u);
  if ((u & 0x7f800000u) == 0x7f800000u) r = u;   // inf/nan: truncate
  return (unsigned short)(r >> 16);
}

// ---------------------------------------------------------------- templated batched WMMA GEMM
// C[b1,b2,m,n] = alpha * sum_k A[b1,b2,m,k] * Bt[b1,b2,n,k] (+bias[n]) (relu?) (+Cin)
// All strides are compile-time so address math folds into immediate offsets.
// Each wave computes a 16 x (16*NT) tile: A fragment reused across NT WMMAs.
// OM: 0 = f32 out, 1 = bf16 out, 2 = f32 out += Cin
template<int NT,
         long long AB1, long long AB2, long long AROW,
         long long B_B1, long long B_B2, long long BROW, long long BK,
         long long CB1, long long CB2, long long CROW,
         int N, int K, int NB2, bool HAS_BIAS, bool RELU, int OM>
__global__ void wmma_gemm(const unsigned short* __restrict__ A,
                          const unsigned short* __restrict__ Bt,
                          const float* __restrict__ bias,
                          float* __restrict__ Cf, unsigned short* __restrict__ Cb,
                          const float* __restrict__ Cin,
                          float alpha, int M, long long total_waves) {
  static_assert((N / 16) % NT == 0, "N tile grouping");
  long long wid = (long long)blockIdx.x * (blockDim.x >> 5) + (threadIdx.x >> 5);
  if (wid >= total_waves) return;                     // wave-uniform: EXEC stays all-1s
  constexpr int NGRP = (N / 16) / NT;
  int mt = M >> 4;
  long long tilesPer = (long long)mt * NGRP;
  long long batch = wid / tilesPer;
  int tile = (int)(wid - batch * tilesPer);
  int tm = tile / NGRP, tg = tile - tm * NGRP;
  long long b1 = batch / NB2, b2 = batch - b1 * NB2;

  int lane = threadIdx.x & 31;
  int r    = lane & 15;
  int hk   = (lane >> 4) << 3;                        // 0 or 8 (k half select)

  const unsigned short* Ab = A + b1*AB1 + b2*AB2 + (long long)(tm*16 + r)*AROW + hk;
  const unsigned short* Bb[NT];
  #pragma unroll
  for (int s = 0; s < NT; ++s)
    Bb[s] = Bt + b1*B_B1 + b2*B_B2 + (long long)((tg*NT + s)*16 + r)*BROW + hk*BK;

  v8f acc[NT] = {};
  for (int k0 = 0; k0 < K; k0 += 32) {
    __builtin_prefetch(Ab + k0 + 128, 0, 1);          // speculative: global_prefetch_b8
    FragAB fa;
    fa.q[0] = *(const uint4*)(Ab + k0);
    fa.q[1] = *(const uint4*)(Ab + k0 + 16);
    FragAB fb[NT];
    #pragma unroll
    for (int s = 0; s < NT; ++s) {
      if constexpr (BK == 1) {
        fb[s].q[0] = *(const uint4*)(Bb[s] + k0);
        fb[s].q[1] = *(const uint4*)(Bb[s] + k0 + 16);
      } else {
        #pragma unroll
        for (int t = 0; t < 8; ++t) {
          fb[s].s[t]     = Bb[s][(long long)(k0 + t)      * BK];
          fb[s].s[8 + t] = Bb[s][(long long)(k0 + 16 + t) * BK];
        }
      }
    }
    #pragma unroll
    for (int s = 0; s < NT; ++s)
      acc[s] = __builtin_amdgcn_wmma_f32_16x16x32_bf16(false, fa.v, false, fb[s].v,
                                                       (short)0, acc[s], false, false);
  }

  int m0 = (lane >> 4) << 3;            // C layout: lanes 16-31 hold M=8..15
  int n  = lane & 15;
  #pragma unroll
  for (int s = 0; s < NT; ++s) {
    int gn = (tg*NT + s)*16 + n;
    #pragma unroll
    for (int g = 0; g < 8; ++g) {
      int gm = tm*16 + m0 + g;
      float v = acc[s][g] * alpha;
      if constexpr (HAS_BIAS) v += bias[gn];
      if constexpr (RELU)     v = fmaxf(v, 0.f);
      long long ci = b1*CB1 + b2*CB2 + (long long)gm*CROW + gn;
      if constexpr (OM == 0)      Cf[ci] = v;
      else if constexpr (OM == 1) Cb[ci] = f32_to_bf16(v);
      else                        Cf[ci] = v + Cin[ci];
    }
  }
}

#define GEMM_LAUNCH(NT_,AB1_,AB2_,AROW_,BB1_,BB2_,BROW_,BK_,CB1_,CB2_,CROW_,N_,K_,NB2_,BIAS_,RELU_,OM_, \
                    Aptr_,Bptr_,biasp_,Cf_,Cb_,Cin_,alpha_,M_,NB1_)                                      \
  do {                                                                                                   \
    long long tw_ = (long long)(NB1_) * (NB2_) * ((M_)/16) * ((N_)/(16*(NT_)));                          \
    int blocks_ = (int)((tw_ + 7) / 8);                                                                  \
    wmma_gemm<NT_,AB1_,AB2_,AROW_,BB1_,BB2_,BROW_,BK_,CB1_,CB2_,CROW_,N_,K_,NB2_,BIAS_,RELU_,OM_>        \
      <<<blocks_, 256, 0, stream>>>(Aptr_, Bptr_, biasp_, Cf_, Cb_, Cin_, alpha_, M_, tw_);              \
  } while (0)

// ---------------------------------------------------------------- LayerNorm row -> bf16 (D=512, wave per row)
__global__ void ln_rows_bf16(const float* __restrict__ src, const float* __restrict__ w,
                             const float* __restrict__ b, unsigned short* __restrict__ dst, int rows) {
  int wid  = blockIdx.x * (blockDim.x >> 5) + (threadIdx.x >> 5);
  int lane = threadIdx.x & 31;
  if (wid >= rows) return;
  const float* rp = src + (size_t)wid * DD;
  float vals[16]; float s = 0.f;
  #pragma unroll
  for (int t = 0; t < 16; ++t) { vals[t] = rp[lane + 32*t]; s += vals[t]; }
  #pragma unroll
  for (int off = 16; off; off >>= 1) s += __shfl_xor(s, off, 32);
  float mu = s * (1.0f / DD);
  float vs = 0.f;
  #pragma unroll
  for (int t = 0; t < 16; ++t) { float d = vals[t] - mu; vs += d*d; }
  #pragma unroll
  for (int off = 16; off; off >>= 1) vs += __shfl_xor(vs, off, 32);
  float inv = rsqrtf(vs * (1.0f / DD) + 1e-5f);
  unsigned short* o = dst + (size_t)wid * DD;
  #pragma unroll
  for (int t = 0; t < 16; ++t) {
    int d = lane + 32*t;
    o[d] = f32_to_bf16((vals[t] - mu) * inv * w[d] + b[d]);
  }
}

// ---------------------------------------------------------------- small helpers
__global__ void conv_b(unsigned short* __restrict__ dst, const float* __restrict__ src, int n) {
  int i = blockIdx.x * 256 + threadIdx.x;
  if (i < n) dst[i] = f32_to_bf16(src[i]);
}
__global__ void convT_b(unsigned short* __restrict__ dst, const float* __restrict__ src, int R, int C) {
  int i = blockIdx.x * 256 + threadIdx.x;
  if (i >= R * C) return;
  int rr = i / C, c = i - rr * C;
  dst[(size_t)c * R + rr] = f32_to_bf16(src[i]);
}
__global__ void init_slots_k(float* __restrict__ slots, const float* __restrict__ mu,
                             const float* __restrict__ ls, const float* __restrict__ noise) {
  int idx = blockIdx.x * 256 + threadIdx.x;          // B*NS*D = 262144
  int d = idx & (DD - 1);
  slots[idx] = mu[d] + __expf(ls[d]) * noise[idx];
}
// softmax over the 16 slots for each (b,h,j); +EPS; in place (fp32)
__global__ void softmax_slots(float* __restrict__ dots) {
  long long idx = (long long)blockIdx.x * 256 + threadIdx.x;   // B*H*N
  if (idx >= (long long)BB * HH * NN) return;
  int j = (int)(idx & (NN - 1));
  long long bh = idx >> 12;
  size_t base = (size_t)bh * (NS * NN) + j;
  float v[16], m = -3.4e38f;
  #pragma unroll
  for (int i = 0; i < NS; ++i) { v[i] = dots[base + (size_t)i * NN]; m = fmaxf(m, v[i]); }
  float s = 0.f;
  #pragma unroll
  for (int i = 0; i < NS; ++i) { v[i] = __expf(v[i] - m); s += v[i]; }
  float inv = 1.f / s;
  #pragma unroll
  for (int i = 0; i < NS; ++i) dots[base + (size_t)i * NN] = v[i] * inv + 1e-8f;
}
__global__ void rowsum_k(const float* __restrict__ attn, float* __restrict__ sums) {
  __shared__ float red[256];
  const float* r = attn + (size_t)blockIdx.x * NN;
  float s = 0.f;
  for (int j = threadIdx.x; j < NN; j += 256) s += r[j];
  red[threadIdx.x] = s; __syncthreads();
  for (int off = 128; off; off >>= 1) {
    if ((int)threadIdx.x < off) red[threadIdx.x] += red[threadIdx.x + off];
    __syncthreads();
  }
  if (threadIdx.x == 0) sums[blockIdx.x] = red[0];
}
__global__ void attn_norm_k(const float* __restrict__ attn, const float* __restrict__ sums,
                            unsigned short* __restrict__ out) {
  long long idx = (long long)blockIdx.x * 256 + threadIdx.x;   // 16777216
  if (idx >= (long long)BB * HH * NS * NN) return;
  out[idx] = f32_to_bf16(attn[idx] / sums[idx >> 12]);
}
__global__ void gru_k(const float* __restrict__ gi, const float* __restrict__ gh,
                      float* __restrict__ slots) {
  int idx = blockIdx.x * 256 + threadIdx.x;          // 262144
  int rr = idx >> 9, d = idx & 511;
  const float* gir = gi + (size_t)rr * (3 * DD);
  const float* ghr = gh + (size_t)rr * (3 * DD);
  float r = 1.f / (1.f + __expf(-(gir[d] + ghr[d])));
  float z = 1.f / (1.f + __expf(-(gir[DD + d] + ghr[DD + d])));
  float n = tanhf(gir[2 * DD + d] + r * ghr[2 * DD + d]);
  slots[idx] = (1.f - z) * n + z * slots[idx];
}
// keep = gumbel-hard over size-2 axis; mask slots in place, emit masked slots as bf16
__global__ void keep_mask(float* __restrict__ slots, const float* __restrict__ Wkeep,
                          const float* __restrict__ g_keep, unsigned short* __restrict__ skb) {
  int wid  = blockIdx.x * 8 + (threadIdx.x >> 5);     // B*NS = 512 waves
  int lane = threadIdx.x & 31;
  if (wid >= BB * NS) return;
  float* row = slots + (size_t)wid * DD;
  float p0 = 0.f, p1 = 0.f;
  #pragma unroll
  for (int t = 0; t < 16; ++t) {
    int d = lane + 32 * t; float s = row[d];
    p0 += s * Wkeep[2 * d]; p1 += s * Wkeep[2 * d + 1];
  }
  #pragma unroll
  for (int off = 16; off; off >>= 1) { p0 += __shfl_xor(p0, off, 32); p1 += __shfl_xor(p1, off, 32); }
  float keep = ((p1 + g_keep[wid * 2 + 1]) > (p0 + g_keep[wid * 2])) ? 1.f : 0.f;
  #pragma unroll
  for (int t = 0; t < 16; ++t) {
    int d = lane + 32 * t; float v = row[d] * keep;
    row[d] = v; skb[(size_t)wid * DD + d] = f32_to_bf16(v);
  }
}
// hard route over slots axis: out[b,j,:] = slots[b, argmax_i(dots2+g), :]
__global__ void route_out(const float* __restrict__ dots2, const float* __restrict__ g_route,
                          const float* __restrict__ slots, float* __restrict__ out) {
  long long wid = (long long)blockIdx.x * 8 + (threadIdx.x >> 5);   // B*N waves
  if (wid >= (long long)BB * NN) return;
  int b = (int)(wid >> 12), j = (int)(wid & (NN - 1));
  int lane = threadIdx.x & 31;
  float v = -3.4e38f; int bi = lane;
  if (lane < NS) {
    size_t p = ((size_t)b * NS + lane) * NN + j;
    v = dots2[p] + g_route[p];
  }
  #pragma unroll
  for (int off = 16; off; off >>= 1) {
    float ov = __shfl_xor(v, off, 32);
    int   oi = __shfl_xor(bi, off, 32);
    if (ov > v || (ov == v && oi < bi)) { v = ov; bi = oi; }
  }
  const float* srow = slots + ((size_t)b * NS + bi) * DD;
  float* orow = out + ((size_t)b * NN + j) * DD;
  #pragma unroll
  for (int t = 0; t < 16; ++t) { int d = lane + 32 * t; orow[d] = srow[d]; }
}

// ---------------------------------------------------------------- host
extern "C" void kernel_launch(void* const* d_in, const int* in_sizes, int n_in,
                              void* d_out, int out_size, void* d_ws, size_t ws_size,
                              hipStream_t stream) {
  const float* inputs   = (const float*)d_in[0];
  const float* slots_mu = (const float*)d_in[1];
  const float* slots_ls = (const float*)d_in[2];
  const float* ln_in_w  = (const float*)d_in[3];
  const float* ln_in_b  = (const float*)d_in[4];
  const float* ln_s_w   = (const float*)d_in[5];
  const float* ln_s_b   = (const float*)d_in[6];
  const float* ln_ff_w  = (const float*)d_in[7];
  const float* ln_ff_b  = (const float*)d_in[8];
  const float* Wq = (const float*)d_in[9];   const float* bq = (const float*)d_in[10];
  const float* Wk = (const float*)d_in[11];  const float* bk = (const float*)d_in[12];
  const float* Wv = (const float*)d_in[13];  const float* bv = (const float*)d_in[14];
  const float* Wc = (const float*)d_in[15];  const float* bc = (const float*)d_in[16];
  const float* Wih = (const float*)d_in[17]; const float* bih = (const float*)d_in[18];
  const float* Whh = (const float*)d_in[19]; const float* bhh = (const float*)d_in[20];
  const float* W1 = (const float*)d_in[21];  const float* b1 = (const float*)d_in[22];
  const float* W2 = (const float*)d_in[23];  const float* b2 = (const float*)d_in[24];
  const float* Wkeep = (const float*)d_in[25];
  const float* slot_noise = (const float*)d_in[26];
  const float* g_keep = (const float*)d_in[27];
  const float* g_route = (const float*)d_in[28];
  float* out = (float*)d_out;

  // workspace carve-up (256B aligned)
  size_t off = 0;
  auto alloc = [&](size_t bytes) {
    off = (off + 255) & ~(size_t)255;
    void* p = (char*)d_ws + off; off += bytes; return p;
  };
  const size_t TOK = (size_t)BB * NN;                 // 131072 rows
  unsigned short* xb    = (unsigned short*)alloc(TOK * DD * 2);
  unsigned short* kbuf  = (unsigned short*)alloc(TOK * DD * 2);
  unsigned short* vbuf  = (unsigned short*)alloc(TOK * DD * 2);
  unsigned short* qb    = (unsigned short*)alloc((size_t)BB * NS * DD * 2);
  float*          dots  = (float*)alloc((size_t)BB * HH * NS * NN * 4);
  unsigned short* attnb = (unsigned short*)alloc((size_t)BB * HH * NS * NN * 2);
  float*          sums  = (float*)alloc((size_t)BB * HH * NS * 4);
  float*          slots = (float*)alloc((size_t)BB * NS * DD * 4);
  unsigned short* sb    = (unsigned short*)alloc((size_t)BB * NS * DD * 2);
  unsigned short* slotsb= (unsigned short*)alloc((size_t)BB * NS * DD * 2);
  unsigned short* sffb  = (unsigned short*)alloc((size_t)BB * NS * DD * 2);
  unsigned short* updb  = (unsigned short*)alloc((size_t)BB * NS * DD * 2);
  unsigned short* updcb = (unsigned short*)alloc((size_t)BB * NS * DD * 2);
  unsigned short* skb   = (unsigned short*)alloc((size_t)BB * NS * DD * 2);
  float*          gi    = (float*)alloc((size_t)BB * NS * 3 * DD * 4);
  float*          gh    = (float*)alloc((size_t)BB * NS * 3 * DD * 4);
  unsigned short* hb    = (unsigned short*)alloc((size_t)BB * NS * HID * 2);
  float*          dots2 = (float*)alloc((size_t)BB * NS * NN * 4);
  unsigned short* WkT = (unsigned short*)alloc((size_t)DD * DD * 2);
  unsigned short* WvT = (unsigned short*)alloc((size_t)DD * DD * 2);
  unsigned short* WqT = (unsigned short*)alloc((size_t)DD * DD * 2);
  unsigned short* WcT = (unsigned short*)alloc((size_t)DD * DD * 2);
  unsigned short* WihB= (unsigned short*)alloc((size_t)3 * DD * DD * 2);
  unsigned short* WhhB= (unsigned short*)alloc((size_t)3 * DD * DD * 2);
  unsigned short* W1T = (unsigned short*)alloc((size_t)DD * HID * 2);
  unsigned short* W2T = (unsigned short*)alloc((size_t)HID * DD * 2);

  // weights -> bf16 (transposed to [N,K] where needed)
  convT_b<<<(DD*DD+255)/256, 256, 0, stream>>>(WkT, Wk, DD, DD);
  convT_b<<<(DD*DD+255)/256, 256, 0, stream>>>(WvT, Wv, DD, DD);
  convT_b<<<(DD*DD+255)/256, 256, 0, stream>>>(WqT, Wq, DD, DD);
  convT_b<<<(DD*DD+255)/256, 256, 0, stream>>>(WcT, Wc, DD, DD);
  conv_b <<<(3*DD*DD+255)/256, 256, 0, stream>>>(WihB, Wih, 3*DD*DD);   // already [3D, D]=[N,K]
  conv_b <<<(3*DD*DD+255)/256, 256, 0, stream>>>(WhhB, Whh, 3*DD*DD);
  convT_b<<<(DD*HID+255)/256, 256, 0, stream>>>(W1T, W1, DD, HID);
  convT_b<<<(HID*DD+255)/256, 256, 0, stream>>>(W2T, W2, HID, DD);

  // x = LN(inputs) -> bf16 ; k = x@Wk+bk ; v = x@Wv+bv (all bf16)
  ln_rows_bf16<<<(int)(TOK/8), 256, 0, stream>>>(inputs, ln_in_w, ln_in_b, xb, (int)TOK);
  GEMM_LAUNCH(4, 0,0,DD, 0,0,DD,1, 0,0,DD, DD,DD,1, true,false,1,
              xb, WkT, bk, nullptr, kbuf, nullptr, 1.f, (int)TOK, 1);
  GEMM_LAUNCH(4, 0,0,DD, 0,0,DD,1, 0,0,DD, DD,DD,1, true,false,1,
              xb, WvT, bv, nullptr, vbuf, nullptr, 1.f, (int)TOK, 1);

  init_slots_k<<<(BB*NS*DD)/256, 256, 0, stream>>>(slots, slots_mu, slots_ls, slot_noise);

  for (int it = 0; it < 3; ++it) {
    ln_rows_bf16<<<64, 256, 0, stream>>>(slots, ln_s_w, ln_s_b, sb, BB*NS);
    // q = LN(slots)@Wq + bq -> bf16, laid out [b,i,h,dh]
    GEMM_LAUNCH(4, 0,0,DD, 0,0,DD,1, 0,0,DD, DD,DD,1, true,false,1,
                sb, WqT, bq, nullptr, qb, nullptr, 1.f, BB*NS, 1);
    // dots[b,h,i,j] = SCALE * q.kT  (batched B,H; M=16,N=4096,K=64)
    GEMM_LAUNCH(4, (long long)NS*DD,DH,DD, (long long)NN*DD,DH,DD,1,
                (long long)HH*NS*NN,(long long)NS*NN,NN, NN,DH,HH, false,false,0,
                qb, kbuf, nullptr, dots, nullptr, nullptr, SCALE, NS, BB);
    softmax_slots<<<(BB*HH*NN)/256, 256, 0, stream>>>(dots);
    rowsum_k<<<BB*HH*NS, 256, 0, stream>>>(dots, sums);
    attn_norm_k<<<(BB*HH*NS*NN)/256, 256, 0, stream>>>(dots, sums, attnb);
    // upd[b,i,h*64+d] = attn . v  (batched B,H; M=16,N=64,K=4096; strided-B path)
    GEMM_LAUNCH(4, (long long)HH*NS*NN,(long long)NS*NN,NN, (long long)NN*DD,DH,1,DD,
                (long long)NS*DD,DH,DD, DH,NN,HH, false,false,1,
                attnb, vbuf, nullptr, nullptr, updb, nullptr, 1.f, NS, BB);
    // updc = upd@Wc + bc -> bf16
    GEMM_LAUNCH(4, 0,0,DD, 0,0,DD,1, 0,0,DD, DD,DD,1, true,false,1,
                updb, WcT, bc, nullptr, updcb, nullptr, 1.f, BB*NS, 1);
    conv_b<<<(BB*NS*DD)/256, 256, 0, stream>>>(slotsb, slots, BB*NS*DD);
    // gi = updc@Wih^T + bih ; gh = slots@Whh^T + bhh (fp32 out)
    GEMM_LAUNCH(4, 0,0,DD, 0,0,DD,1, 0,0,3*DD, 3*DD,DD,1, true,false,0,
                updcb, WihB, bih, gi, nullptr, nullptr, 1.f, BB*NS, 1);
    GEMM_LAUNCH(4, 0,0,DD, 0,0,DD,1, 0,0,3*DD, 3*DD,DD,1, true,false,0,
                slotsb, WhhB, bhh, gh, nullptr, nullptr, 1.f, BB*NS, 1);
    gru_k<<<(BB*NS*DD)/256, 256, 0, stream>>>(gi, gh, slots);
    // MLP residual
    ln_rows_bf16<<<64, 256, 0, stream>>>(slots, ln_ff_w, ln_ff_b, sffb, BB*NS);
    GEMM_LAUNCH(4, 0,0,DD, 0,0,DD,1, 0,0,HID, HID,DD,1, true,true,1,
                sffb, W1T, b1, nullptr, hb, nullptr, 1.f, BB*NS, 1);
    GEMM_LAUNCH(4, 0,0,HID, 0,0,HID,1, 0,0,DD, DD,HID,1, true,false,2,
                hb, W2T, b2, slots, nullptr, slots, 1.f, BB*NS, 1);
  }

  // keep gate + mask; dots2 = SCALE * slots.xT (batched over B); hard route gather
  keep_mask<<<64, 256, 0, stream>>>(slots, Wkeep, g_keep, skb);
  GEMM_LAUNCH(4, (long long)NS*DD,0,DD, (long long)NN*DD,0,DD,1,
              (long long)NS*NN,0,NN, NN,DD,1, false,false,0,
              skb, xb, nullptr, dots2, nullptr, nullptr, SCALE, NS, BB);
  route_out<<<(int)(TOK/8), 256, 0, stream>>>(dots2, g_route, slots, out);
}
